// BiTrapSGlow_76845554860465
// MI455X (gfx1250) — matrix operand order repaired and checked by reading, not compiled, not run on hardware
//
#include <hip/hip_runtime.h>
#include <stdint.h>

// ---------------- problem constants ----------------
#define DMODEL 1024
#define DFF    4096
#define NHEAD  16
#define DHEAD  64
#define SEQ    1024
#define NBATCH 4
#define NROWS  (NBATCH * SEQ)   // 4096

typedef __attribute__((ext_vector_type(16))) __bf16 v16bf;
typedef __attribute__((ext_vector_type(8)))  float  v8f;
typedef __attribute__((ext_vector_type(4)))  unsigned int v4u;
typedef __attribute__((ext_vector_type(8)))  int    v8i;
typedef __attribute__((ext_vector_type(4)))  int    v4i;
typedef unsigned short u16t;

// ---------------- bf16 helpers ----------------
__device__ __forceinline__ u16t f2bf(float f) {
  unsigned int u = __float_as_uint(f);
  u += 0x7FFFu + ((u >> 16) & 1u);              // round-to-nearest-even
  return (u16t)(u >> 16);
}

union Frag16 { v16bf bf; uint4 u[2]; };

// A-fragment (16x32 16-bit, ISA 7.12.2): M = lane&15.
// elems 0..7  -> K = 8*half + {0..7}; elems 8..15 -> K = 16 + 8*half + {0..7}
__device__ __forceinline__ v16bf load_afrag(const u16t* base, int row_stride, int lane) {
  int half = lane >> 4;
  int m    = lane & 15;
  const u16t* p = base + (size_t)m * row_stride + half * 8;
  Frag16 f;
  f.u[0] = *(const uint4*)(p);
  f.u[1] = *(const uint4*)(p + 16);
  return f.bf;
}

// B-fragment (32x16 16-bit), memory as [N][K] rows:
// N = lane&15; elems 0..7 -> K = 8*half + {0..7}; elems 8..15 -> K = 16 + 8*half + {0..7}
__device__ __forceinline__ v16bf load_bfrag(const u16t* base, int row_stride, int lane) {
  int half = lane >> 4;
  int n    = lane & 15;
  const u16t* p = base + (size_t)n * row_stride + half * 8;
  Frag16 f;
  f.u[0] = *(const uint4*)(p);
  f.u[1] = *(const uint4*)(p + 16);
  return f.bf;
}

#define WMMA_BF16(A, B, C) \
  __builtin_amdgcn_wmma_f32_16x16x32_bf16(false, (A), false, (B), (short)0, (C), false, false)

// ---------------- TDM: 2-D bf16 tile (tile_k x tile_n rows) global -> LDS ----------------
// D# packing per CDNA5 ISA. Row-by-row contiguous LDS write, row
// length = tile_k elements -> matches load_bfrag(row_stride = tile_k).
__device__ __forceinline__ void tdm_load_tile_b16(unsigned lds_addr, const void* gptr,
                                                  unsigned tile_k, unsigned tile_n,
                                                  unsigned tensor_k,
                                                  unsigned long long row_stride_elems) {
  unsigned long long ga = (unsigned long long)(uintptr_t)gptr;
  v4u g0;
  g0[0] = 1u;                                            // count=1 (valid), user mode
  g0[1] = lds_addr;                                      // lds_addr [63:32]
  g0[2] = (unsigned)(ga & 0xFFFFFFFFu);                  // global_addr [95:64]
  g0[3] = (unsigned)((ga >> 32) & 0x01FFFFFFu) | (2u << 30); // global_addr[56:32] | type=2

  unsigned td0 = tensor_k;         // tensor_dim0 (elements, dim0 = contiguous K)
  unsigned td1 = tile_n;           // tensor_dim1 (rows)
  unsigned long long s0 = row_stride_elems;              // tensor_dim0_stride
  v8i g1;
  g1[0] = (int)(1u << 16);                               // workgroup_mask=0, data_size=1 (2B)
  g1[1] = (int)((td0 & 0xFFFFu) << 16);                  // tensor_dim0[15:0] @ bits 63:48
  g1[2] = (int)(((td0 >> 16) & 0xFFFFu) | ((td1 & 0xFFFFu) << 16));
  g1[3] = (int)(((td1 >> 16) & 0xFFFFu) | ((tile_k & 0xFFFFu) << 16)); // tile_dim0
  g1[4] = (int)(tile_n & 0xFFFFu);                       // tile_dim1 (tile_dim2 = 0)
  g1[5] = (int)(unsigned)(s0 & 0xFFFFFFFFu);             // tensor_dim0_stride[31:0]
  g1[6] = (int)(unsigned)((s0 >> 32) & 0xFFFFu);         // stride[47:32]; dim1_stride=0
  g1[7] = 0;

  v4i gz = {0, 0, 0, 0};
#if defined(__clang_major__) && (__clang_major__ >= 23)
  v8i gz8 = {0, 0, 0, 0, 0, 0, 0, 0};
  __builtin_amdgcn_tensor_load_to_lds(g0, g1, gz, gz, gz8, 0);
#else
  __builtin_amdgcn_tensor_load_to_lds(g0, g1, gz, gz, 0);
#endif
}

// ---------------- weight transpose + convert: W[K,N] f32 -> Wt[N,K] bf16 ----------------
__global__ void wtrans_kernel(const float* __restrict__ W, u16t* __restrict__ Wt,
                              int K, int N) {
  size_t idx = (size_t)blockIdx.x * blockDim.x + threadIdx.x;
  if (idx >= (size_t)K * N) return;
  int n  = (int)(idx / K);
  int kk = (int)(idx % K);
  Wt[idx] = f2bf(W[(size_t)kk * N + n]);
}

// ---------------- LayerNorm (f32 in) -> bf16 out, one block per row ----------------
__global__ void ln_bf16_kernel(const float* __restrict__ x, const float* __restrict__ g,
                               const float* __restrict__ bta, u16t* __restrict__ y) {
  __shared__ float red[256];
  int row = blockIdx.x;
  int tid = threadIdx.x;
  const float4 xv = *(const float4*)(x + (size_t)row * DMODEL + tid * 4);

  float s = xv.x + xv.y + xv.z + xv.w;
  red[tid] = s; __syncthreads();
  for (int off = 128; off > 0; off >>= 1) { if (tid < off) red[tid] += red[tid + off]; __syncthreads(); }
  float mean = red[0] * (1.0f / DMODEL);
  __syncthreads();

  float d0 = xv.x - mean, d1 = xv.y - mean, d2 = xv.z - mean, d3 = xv.w - mean;
  red[tid] = d0 * d0 + d1 * d1 + d2 * d2 + d3 * d3; __syncthreads();
  for (int off = 128; off > 0; off >>= 1) { if (tid < off) red[tid] += red[tid + off]; __syncthreads(); }
  float rstd = rsqrtf(red[0] * (1.0f / DMODEL) + 1e-5f);

  const float4 gv = *(const float4*)(g   + tid * 4);
  const float4 bv = *(const float4*)(bta + tid * 4);
  u16t o0 = f2bf(d0 * rstd * gv.x + bv.x);
  u16t o1 = f2bf(d1 * rstd * gv.y + bv.y);
  u16t o2 = f2bf(d2 * rstd * gv.z + bv.z);
  u16t o3 = f2bf(d3 * rstd * gv.w + bv.w);
  uint2 pack;
  pack.x = (unsigned)o0 | ((unsigned)o1 << 16);
  pack.y = (unsigned)o2 | ((unsigned)o3 << 16);
  *(uint2*)(y + (size_t)row * DMODEL + tid * 4) = pack;
}

// ---------------- TDM-staged bf16 GEMM: C = A[M,K] @ Wt[N,K]^T ----------------
// Block = 8 waves covering 256 rows x 64 cols; all waves share the N-tile.
// Wave 0 streams the shared 64x32 B tile into double-buffered LDS via the
// Tensor Data Mover (TENSORcnt) while everyone computes from the other buffer.
// EPI 0: (acc+bias)*alpha -> bf16 scatter [B,H,S,d]    (Q alpha=1/8, K alpha=1)
// EPI 1: (acc+bias)       -> bf16 scatter [B,H,d,S]    (V transposed)
// EPI 2: gelu(acc+bias)   -> bf16 row-major [M,N]      (W1)
// EPI 3: resid+acc+bias   -> f32 row-major [M,N]       (Wo, W2)
template <int EPI>
__global__ void gemm_bf16_kernel(const u16t* __restrict__ A, const u16t* __restrict__ Bw,
                                 const float* __restrict__ bias,
                                 u16t* __restrict__ outb, float* __restrict__ outf,
                                 const float* __restrict__ resid,
                                 int M, int N, int K, float alpha) {
  __shared__ __align__(16) u16t Bs[2][64 * 32];   // double-buffered B tile, 8 KB
  int lane = threadIdx.x & 31;
  int wave = threadIdx.x >> 5;
  int nblk = N >> 6;
  int bM = blockIdx.x / nblk, bN = blockIdx.x % nblk;
  int m0 = bM * 256 + wave * 32;                  // grid sized exactly: m0 < M always
  int n0 = bN * 64;

  v8f z = {};
  v8f acc[2][4];
  for (int i = 0; i < 2; ++i)
    for (int j = 0; j < 4; ++j) acc[i][j] = z;

  if (wave == 0)
    tdm_load_tile_b16((unsigned)(uintptr_t)&Bs[0][0], Bw + (size_t)n0 * K,
                      32, 64, (unsigned)K, (unsigned long long)K);

  for (int kc = 0; kc < K; kc += 32) {
    int cur = (kc >> 5) & 1;
    if (wave == 0) __builtin_amdgcn_s_wait_tensorcnt(0);   // buffer `cur` landed
    __syncthreads();                                        // ready + prev reads done
    if (wave == 0 && kc + 32 < K)
      tdm_load_tile_b16((unsigned)(uintptr_t)&Bs[cur ^ 1][0],
                        Bw + (size_t)n0 * K + kc + 32,
                        32, 64, (unsigned)K, (unsigned long long)K);
    if (kc + 32 < K)
      __builtin_prefetch(A + (size_t)m0 * K + kc + 32, 0, 0);  // global_prefetch_b8

    v16bf a0 = load_afrag(A + (size_t)m0 * K + kc, K, lane);
    v16bf a1 = load_afrag(A + (size_t)(m0 + 16) * K + kc, K, lane);
#pragma unroll
    for (int j = 0; j < 4; ++j) {
      v16bf b = load_bfrag(&Bs[cur][j * 16 * 32], 32, lane);  // ds_load_b128
      acc[0][j] = WMMA_BF16(a0, b, acc[0][j]);
      acc[1][j] = WMMA_BF16(a1, b, acc[1][j]);
    }
  }

  int half = lane >> 4;
  int nlo  = lane & 15;
#pragma unroll
  for (int i = 0; i < 2; ++i) {
#pragma unroll
    for (int j = 0; j < 4; ++j) {
      int col = n0 + j * 16 + nlo;
      float bb = bias[col];
#pragma unroll
      for (int e = 0; e < 8; ++e) {
        int row = m0 + i * 16 + e + half * 8;      // C layout: M = vgpr + 8*(lane>=16)
        float v = acc[i][j][e] + bb;
        if (EPI == 0) {                            // scatter [B,H,S,d] with scale
          int b = row >> 10, s = row & 1023;
          int h = col >> 6,  dd = col & 63;
          outb[(((size_t)(b * NHEAD + h) * SEQ + s) * DHEAD) + dd] = f2bf(v * alpha);
        } else if (EPI == 1) {                     // scatter [B,H,d,S] (transposed V)
          int b = row >> 10, s = row & 1023;
          int h = col >> 6,  dd = col & 63;
          outb[(((size_t)(b * NHEAD + h) * DHEAD + dd) * SEQ) + s] = f2bf(v);
        } else if (EPI == 2) {                     // exact GELU
          float g = 0.5f * v * (1.0f + erff(v * 0.70710678118654752f));
          outb[(size_t)row * N + col] = f2bf(g);
        } else {                                   // residual add, f32
          outf[(size_t)row * N + col] = resid[(size_t)row * N + col] + v;
        }
      }
    }
  }
}

// ---------------- fused flash attention: one wave per 16-row q tile ----------------
__global__ void attn_kernel(const u16t* __restrict__ q,   // [B,H,S,d] (pre-scaled)
                            const u16t* __restrict__ k,   // [B,H,S,d]
                            const u16t* __restrict__ vt,  // [B,H,d,S]
                            const float* __restrict__ attn_bias, // [B-1,H]
                            const float* __restrict__ mask,      // [B,1,S,S]
                            u16t* __restrict__ ctx) {            // [B,S,D] bf16
  __shared__ __align__(16) u16t P[8][16 * 32];   // per-wave P tile (16 q x 32 keys)
  int lane = threadIdx.x & 31;
  int wave = threadIdx.x >> 5;
  int gw = blockIdx.x * 8 + wave;                // over B*H*(S/16) = 4096 waves
  int qt = gw & 63;
  int bh = gw >> 6;
  int h = bh & (NHEAD - 1);
  int b = bh >> 4;
  int q0 = qt << 4;

  const u16t* qbase = q  + (size_t)bh * SEQ * DHEAD + (size_t)q0 * DHEAD;
  const u16t* kbase = k  + (size_t)bh * SEQ * DHEAD;
  const u16t* vbase = vt + (size_t)bh * DHEAD * SEQ;
  const float* mbase = mask + (size_t)b * SEQ * SEQ;
  float biasb = (b > 0) ? attn_bias[(b - 1) * NHEAD + h] : 0.0f;

  int half = lane >> 4;
  int nlo  = lane & 15;

  v16bf aq0 = load_afrag(qbase + 0,  DHEAD, lane);   // q tile, K = 0..31
  v16bf aq1 = load_afrag(qbase + 32, DHEAD, lane);   // q tile, K = 32..63

  float mrow[8], lrow[8];
#pragma unroll
  for (int e = 0; e < 8; ++e) { mrow[e] = -3.0e38f; lrow[e] = 0.0f; }
  v8f z = {};
  v8f acc[4];
#pragma unroll
  for (int j = 0; j < 4; ++j) acc[j] = z;

  for (int kt = 0; kt < SEQ; kt += 32) {
    // scores for keys kt..kt+15 and kt+16..kt+31 (each: 2 WMMA over d=64)
    v8f s0 = z, s1 = z;
    {
      v16bf b0 = load_bfrag(kbase + (size_t)kt * DHEAD + 0,  DHEAD, lane);
      v16bf b1 = load_bfrag(kbase + (size_t)kt * DHEAD + 32, DHEAD, lane);
      s0 = WMMA_BF16(aq0, b0, s0);
      s0 = WMMA_BF16(aq1, b1, s0);
    }
    {
      v16bf b0 = load_bfrag(kbase + (size_t)(kt + 16) * DHEAD + 0,  DHEAD, lane);
      v16bf b1 = load_bfrag(kbase + (size_t)(kt + 16) * DHEAD + 32, DHEAD, lane);
      s1 = WMMA_BF16(aq0, b0, s1);
      s1 = WMMA_BF16(aq1, b1, s1);
    }
    // bias + mask + online softmax (rows in (vgpr e, half); cols across 16 lanes)
#pragma unroll
    for (int e = 0; e < 8; ++e) {
      int row = q0 + e + half * 8;
      float mk0 = mbase[(size_t)row * SEQ + kt + nlo];
      float mk1 = mbase[(size_t)row * SEQ + kt + 16 + nlo];
      float p0 = (s0[e] + biasb) * mk0 + (mk0 > 0.0f ? 0.0f : -1e9f);
      float p1 = (s1[e] + biasb) * mk1 + (mk1 > 0.0f ? 0.0f : -1e9f);

      float mx = fmaxf(p0, p1);
      for (int w = 1; w < 16; w <<= 1) mx = fmaxf(mx, __shfl_xor(mx, w, 32));
      float mnew  = fmaxf(mrow[e], mx);
      float scale = __expf(mrow[e] - mnew);
      float e0 = __expf(p0 - mnew);
      float e1 = __expf(p1 - mnew);
      float ps = e0 + e1;
      for (int w = 1; w < 16; w <<= 1) ps += __shfl_xor(ps, w, 32);
      lrow[e] = lrow[e] * scale + ps;
      mrow[e] = mnew;
#pragma unroll
      for (int j = 0; j < 4; ++j) acc[j][e] *= scale;

      int r = e + half * 8;                       // C-layout -> LDS row-major P tile
      P[wave][r * 32 + nlo]      = f2bf(e0);
      P[wave][r * 32 + 16 + nlo] = f2bf(e1);
    }
    asm volatile("s_wait_dscnt 0" ::: "memory");  // LDS RAW: P stores -> A-frag loads
    v16bf ap = load_afrag(&P[wave][0], 32, lane); // P as A (16 x 32 keys)
#pragma unroll
    for (int j = 0; j < 4; ++j) {                 // ctx += P @ V (V transposed: contiguous B)
      v16bf bv = load_bfrag(vbase + (size_t)(j * 16) * SEQ + kt, SEQ, lane);
      acc[j] = WMMA_BF16(ap, bv, acc[j]);
    }
  }

  // normalize and scatter to [B,S,D] bf16
#pragma unroll
  for (int e = 0; e < 8; ++e) {
    float inv = 1.0f / lrow[e];
    int row = q0 + e + half * 8;
    size_t ob = ((size_t)b * SEQ + row) * DMODEL + h * DHEAD;
#pragma unroll
    for (int j = 0; j < 4; ++j)
      ctx[ob + j * 16 + nlo] = f2bf(acc[j][e] * inv);
  }
}

// ---------------- host-side launch ----------------
extern "C" void kernel_launch(void* const* d_in, const int* in_sizes, int n_in,
                              void* d_out, int out_size, void* d_ws, size_t ws_size,
                              hipStream_t stream) {
  const float* x     = (const float*)d_in[0];
  const float* abias = (const float*)d_in[1];
  const float* mask  = (const float*)d_in[2];
  const float* ln1g  = (const float*)d_in[3];
  const float* ln1b  = (const float*)d_in[4];
  const float* Wq = (const float*)d_in[5];  const float* bq = (const float*)d_in[6];
  const float* Wk = (const float*)d_in[7];  const float* bk = (const float*)d_in[8];
  const float* Wv = (const float*)d_in[9];  const float* bv = (const float*)d_in[10];
  const float* Wo = (const float*)d_in[11]; const float* bo = (const float*)d_in[12];
  const float* ln2g = (const float*)d_in[13];
  const float* ln2b = (const float*)d_in[14];
  const float* W1 = (const float*)d_in[15]; const float* b1 = (const float*)d_in[16];
  const float* W2 = (const float*)d_in[17]; const float* b2 = (const float*)d_in[18];
  float* out = (float*)d_out;

  char* ws = (char*)d_ws;
  size_t off = 0;
  auto alloc = [&](size_t bytes) -> void* {
    void* p = ws + off;
    off += (bytes + 255) & ~(size_t)255;
    return p;
  };
  u16t* WqT = (u16t*)alloc((size_t)DMODEL * DMODEL * 2);
  u16t* WkT = (u16t*)alloc((size_t)DMODEL * DMODEL * 2);
  u16t* WvT = (u16t*)alloc((size_t)DMODEL * DMODEL * 2);
  u16t* WoT = (u16t*)alloc((size_t)DMODEL * DMODEL * 2);
  u16t* W1T = (u16t*)alloc((size_t)DMODEL * DFF * 2);
  u16t* W2T = (u16t*)alloc((size_t)DFF * DMODEL * 2);
  u16t* y1  = (u16t*)alloc((size_t)NROWS * DMODEL * 2);
  u16t* qb  = (u16t*)alloc((size_t)NROWS * DMODEL * 2);
  u16t* kb  = (u16t*)alloc((size_t)NROWS * DMODEL * 2);
  u16t* vtb = (u16t*)alloc((size_t)NROWS * DMODEL * 2);
  u16t* ctx = (u16t*)alloc((size_t)NROWS * DMODEL * 2);
  float* x1 = (float*)alloc((size_t)NROWS * DMODEL * 4);
  u16t* y2  = (u16t*)alloc((size_t)NROWS * DMODEL * 2);
  u16t* hb  = (u16t*)alloc((size_t)NROWS * DFF * 2);

  const int T = 256;
  // weight transpose + bf16 convert
  wtrans_kernel<<<(DMODEL * DMODEL + T - 1) / T, T, 0, stream>>>(Wq, WqT, DMODEL, DMODEL);
  wtrans_kernel<<<(DMODEL * DMODEL + T - 1) / T, T, 0, stream>>>(Wk, WkT, DMODEL, DMODEL);
  wtrans_kernel<<<(DMODEL * DMODEL + T - 1) / T, T, 0, stream>>>(Wv, WvT, DMODEL, DMODEL);
  wtrans_kernel<<<(DMODEL * DMODEL + T - 1) / T, T, 0, stream>>>(Wo, WoT, DMODEL, DMODEL);
  wtrans_kernel<<<(DMODEL * DFF + T - 1) / T, T, 0, stream>>>(W1, W1T, DMODEL, DFF);
  wtrans_kernel<<<(DFF * DMODEL + T - 1) / T, T, 0, stream>>>(W2, W2T, DFF, DMODEL);

  // LN1
  ln_bf16_kernel<<<NROWS, 256, 0, stream>>>(x, ln1g, ln1b, y1);

  // QKV projections: blocks = (M/256)*(N/64) = 16*16 = 256
  gemm_bf16_kernel<0><<<256, 256, 0, stream>>>(y1, WqT, bq, qb, nullptr, nullptr,
                                               NROWS, DMODEL, DMODEL, 0.125f);
  gemm_bf16_kernel<0><<<256, 256, 0, stream>>>(y1, WkT, bk, kb, nullptr, nullptr,
                                               NROWS, DMODEL, DMODEL, 1.0f);
  gemm_bf16_kernel<1><<<256, 256, 0, stream>>>(y1, WvT, bv, vtb, nullptr, nullptr,
                                               NROWS, DMODEL, DMODEL, 1.0f);

  // fused attention: B*H*(S/16) = 4096 waves -> 512 blocks of 8 waves
  attn_kernel<<<512, 256, 0, stream>>>(qb, kb, vtb, abias, mask, ctx);

  // O projection + residual (f32)
  gemm_bf16_kernel<3><<<256, 256, 0, stream>>>(ctx, WoT, bo, nullptr, x1, x,
                                               NROWS, DMODEL, DMODEL, 1.0f);
  // LN2
  ln_bf16_kernel<<<NROWS, 256, 0, stream>>>(x1, ln2g, ln2b, y2);

  // FFN up + GELU: blocks = 16*64 = 1024
  gemm_bf16_kernel<2><<<1024, 256, 0, stream>>>(y2, W1T, b1, hb, nullptr, nullptr,
                                                NROWS, DFF, DMODEL, 1.0f);
  // FFN down + residual -> d_out: blocks = 16*16 = 256
  gemm_bf16_kernel<3><<<256, 256, 0, stream>>>(hb, W2T, b2, nullptr, out, x1,
                                               NROWS, DMODEL, DFF, 1.0f);
}
